// STSelfAttention_19172734009887
// MI455X (gfx1250) — compile-verified, matrix operand order checked by reading, not run on hardware
//
#include <hip/hip_runtime.h>
#include <hip/hip_bf16.h>

// ---------------------------------------------------------------------------
// Types for CDNA5 WMMA (gfx1250, wave32): D(f32 16x16) = A(bf16 16x32) x B(bf16 32x16) + C
// ---------------------------------------------------------------------------
typedef __attribute__((ext_vector_type(16))) __bf16 v16bf;
typedef __attribute__((ext_vector_type(8)))  float  v8f;

union U16 {
    v16bf v;
    unsigned short h[16];
    unsigned int   u[8];
};

// K index pattern of the 16-bit A operand (and symmetric B operand):
// lane half h (lane>>4), VGPR v -> first K of the bf16 pair.
#define KP(v, h) ((((v) >> 2) << 4) + (h) * 8 + (((v) & 3) << 1))

#define NEGINF (-__builtin_inff())

__device__ __forceinline__ unsigned short f2bf(float f) {
    unsigned int u = __float_as_uint(f);
    unsigned int r = u + 0x7fffu + ((u >> 16) & 1u);   // round-to-nearest-even-ish
    return (unsigned short)(r >> 16);
}
__device__ __forceinline__ unsigned int pack2(float lo, float hi) {
    return ((unsigned int)f2bf(hi) << 16) | (unsigned int)f2bf(lo);
}

__device__ __forceinline__ v8f wmma_bf16(const U16& a, const U16& b, v8f c) {
    return __builtin_amdgcn_wmma_f32_16x16x32_bf16(false, a.v, false, b.v,
                                                   (short)0, c, false, false);
}

// Row reductions across the 16 "column" lanes of each half-wave (C layout cols).
__device__ __forceinline__ float rmax16(float x) {
    x = fmaxf(x, __shfl_xor(x, 1, 16));
    x = fmaxf(x, __shfl_xor(x, 2, 16));
    x = fmaxf(x, __shfl_xor(x, 4, 16));
    x = fmaxf(x, __shfl_xor(x, 8, 16));
    return x;
}
__device__ __forceinline__ float rsum16(float x) {
    x += __shfl_xor(x, 1, 16);
    x += __shfl_xor(x, 2, 16);
    x += __shfl_xor(x, 4, 16);
    x += __shfl_xor(x, 8, 16);
    return x;
}

// Load a 16x16 bf16 tile (row-major, ld=16) as an A/B operand with K padded 16->32.
__device__ __forceinline__ void loadT16pad(U16& d, const unsigned short* __restrict__ base,
                                           int lane) {
    const int mm = lane & 15, h = lane >> 4;
#pragma unroll
    for (int v = 0; v < 8; ++v) {
        d.u[v] = (v < 4) ? *(const unsigned int*)(base + mm * 16 + h * 8 + ((v & 3) << 1))
                         : 0u;
    }
}

// Load one 16x32 B operand from the bf16 weight matrix (ld=128), n-rows at rowbase.
__device__ __forceinline__ void loadW(U16& d, const unsigned short* __restrict__ W,
                                      int rowbase, int ks, int lane) {
    const int nn = lane & 15, h = lane >> 4;
#pragma unroll
    for (int v = 0; v < 8; ++v)
        d.u[v] = *(const unsigned int*)(W + (size_t)(rowbase + nn) * 128 + ks * 32 + KP(v, h));
}

// ---------------------------------------------------------------------------
// Problem constants
// ---------------------------------------------------------------------------
#define Bc 8
#define Tc 12
#define Nc 512
#define Dc 128
#define Kc 16
#define SCALE 0.25f
#define BTN (Bc * Tc * Nc)          // 49152

// Workspace layout (units: unsigned short / bf16 elements)
static constexpr size_t OFF_W   = 0;                         // 576 x 128 bf16 weights
static constexpr size_t NW      = (size_t)576 * 128;         // 73728
static constexpr size_t OFF_PK  = NW;                        // pk   16x64
static constexpr size_t OFF_PVT = OFF_PK + 16 * 64;          // pv^T 64x16
static constexpr size_t OFF_TQ  = OFF_PVT + 16 * 64;         // 75776 (16B aligned in bytes)
static constexpr size_t TSZ = (size_t)Bc * Nc * 2 * 16 * 16; // temporal, T padded to 16
static constexpr size_t OFF_TK = OFF_TQ + TSZ;
static constexpr size_t OFF_TV = OFF_TK + TSZ;
static constexpr size_t GSZ = (size_t)Bc * Tc * 4 * Nc * 16;
static constexpr size_t OFF_GQ = OFF_TV + TSZ;
static constexpr size_t OFF_GK = OFF_GQ + GSZ;
static constexpr size_t OFF_GV = OFF_GK + GSZ;
static constexpr size_t SSZ = (size_t)Bc * Tc * 2 * Nc * 16;
static constexpr size_t OFF_SQ = OFF_GV + GSZ;
static constexpr size_t OFF_SK = OFF_SQ + SSZ;
static constexpr size_t OFF_SV = OFF_SK + SSZ;
static constexpr size_t OFF_CAT = OFF_SV + SSZ;              // BTN x 128 bf16

// ---------------------------------------------------------------------------
// Kernel 0a: convert all projection weights into one bf16 matrix, rows:
//   [0 tq][32 tk][64 tv][96 gq][160 gk][224 gv][288 sq][320 sk][352 sv][384 patq][448 proj]
// ---------------------------------------------------------------------------
__global__ void k0_weights(unsigned short* __restrict__ Wall,
                           const float* tq, const float* tk, const float* tv,
                           const float* gq, const float* gk, const float* gv,
                           const float* sq, const float* sk, const float* sv,
                           const float* pq, const float* pw) {
    int i = blockIdx.x * blockDim.x + threadIdx.x;
    if (i >= (int)NW) return;
    int row = i >> 7, col = i & 127;
    const float* s;
    if      (row <  32) s = tq + (size_t)row * 128;
    else if (row <  64) s = tk + (size_t)(row - 32) * 128;
    else if (row <  96) s = tv + (size_t)(row - 64) * 128;
    else if (row < 160) s = gq + (size_t)(row - 96) * 128;
    else if (row < 224) s = gk + (size_t)(row - 160) * 128;
    else if (row < 288) s = gv + (size_t)(row - 224) * 128;
    else if (row < 320) s = sq + (size_t)(row - 288) * 128;
    else if (row < 352) s = sk + (size_t)(row - 320) * 128;
    else if (row < 384) s = sv + (size_t)(row - 352) * 128;
    else if (row < 448) s = pq + (size_t)(row - 384) * 128;
    else                s = pw + (size_t)(row - 448) * 128;
    Wall[i] = f2bf(s[col]);
}

// Kernel 0b: pk = keys@kw^T + kb (16x64, row-major), pvT = (keys@vw^T + vb)^T (64x16)
__global__ void k0_pkpv(unsigned short* __restrict__ PKo, unsigned short* __restrict__ PVT,
                        const float* __restrict__ pkeys,
                        const float* __restrict__ kw, const float* __restrict__ kb,
                        const float* __restrict__ vw, const float* __restrict__ vb) {
    int tid = threadIdx.x;
#pragma unroll
    for (int j = 0; j < 4; ++j) {
        int idx = tid + j * 256;            // 0..1023
        int c = idx >> 6, e = idx & 63;
        float aK = kb[e], aV = vb[e];
        for (int d = 0; d < 128; ++d) {
            float p = pkeys[c * 128 + d];
            aK += p * kw[e * 128 + d];
            aV += p * vw[e * 128 + d];
        }
        PKo[c * 64 + e] = f2bf(aK);
        PVT[e * 16 + c] = f2bf(aV);
    }
}

// Kernel 0c: zero the T-padded temporal staging buffers (rows t=12..15 stay 0)
__global__ void k0_clear(float4* __restrict__ p, int n) {
    int i = blockIdx.x * blockDim.x + threadIdx.x;
    if (i < n) p[i] = make_float4(0.f, 0.f, 0.f, 0.f);
}

// ---------------------------------------------------------------------------
// Kernel 1: all 9 projections (+ pattern attention fused into geo_k), 16 rows/wave.
// ---------------------------------------------------------------------------
__global__ __launch_bounds__(32) void k1_proj(
    const float* __restrict__ x, const float* __restrict__ xp,
    const unsigned short* __restrict__ Wall,
    const unsigned short* __restrict__ PK, const unsigned short* __restrict__ PVT,
    const float* __restrict__ pat_qb,
    unsigned short* __restrict__ TQ, unsigned short* __restrict__ TK,
    unsigned short* __restrict__ TV, unsigned short* __restrict__ GQ,
    unsigned short* __restrict__ GK, unsigned short* __restrict__ GV,
    unsigned short* __restrict__ SQ, unsigned short* __restrict__ SK,
    unsigned short* __restrict__ SV) {
    __shared__ __align__(16) unsigned short sb[16 * 64];

    const int lane = threadIdx.x;
    const int m = lane & 15, hf = lane >> 4;
    const int row0 = blockIdx.x * 16;

    // A operands: x rows and x_patterns rows (f32 -> bf16 on the fly), K=128 in 4 steps.
    U16 ax[4], ap[4];
#pragma unroll
    for (int ks = 0; ks < 4; ++ks) {
#pragma unroll
        for (int v = 0; v < 8; ++v) {
            int k = ks * 32 + KP(v, hf);
            const float* xr = x + (size_t)(row0 + m) * 128 + k;
            ax[ks].u[v] = pack2(xr[0], xr[1]);
            const float* pr = xp + (size_t)(row0 + m) * 128 + k;
            ap[ks].u[v] = pack2(pr[0], pr[1]);
        }
    }

    // ---- pattern branch: pq = xp @ patqw^T + b  (4 tiles of GD=64) ----
    v8f pq[4];
#pragma unroll
    for (int t4 = 0; t4 < 4; ++t4) {
        v8f c = {};
#pragma unroll
        for (int ks = 0; ks < 4; ++ks) {
            U16 bb; loadW(bb, Wall, 384 + t4 * 16, ks, lane);
            c = wmma_bf16(ap[ks], bb, c);
        }
        float bia = pat_qb[t4 * 16 + m];
#pragma unroll
        for (int i = 0; i < 8; ++i) c[i] += bia;
        pq[t4] = c;
    }
    // C-layout -> A-layout via LDS (16x64 bf16)
#pragma unroll
    for (int t4 = 0; t4 < 4; ++t4)
#pragma unroll
        for (int i = 0; i < 8; ++i)
            sb[(i + hf * 8) * 64 + t4 * 16 + m] = f2bf(pq[t4][i]);
    __syncthreads();
    U16 pqa[2];
#pragma unroll
    for (int ks = 0; ks < 2; ++ks)
#pragma unroll
        for (int v = 0; v < 8; ++v)
            pqa[ks].u[v] = *(const unsigned int*)(sb + m * 64 + ks * 32 + KP(v, hf));
    __syncthreads();

    // scores = pq @ pk^T (16 rows x 16 clusters)
    v8f sc = {};
#pragma unroll
    for (int ks = 0; ks < 2; ++ks) {
        U16 bb;
#pragma unroll
        for (int v = 0; v < 8; ++v)
            bb.u[v] = *(const unsigned int*)(PK + m * 64 + ks * 32 + KP(v, hf));
        sc = wmma_bf16(pqa[ks], bb, sc);
    }
    v8f prb;
#pragma unroll
    for (int i = 0; i < 8; ++i) {
        float s = sc[i] * SCALE;
        float mx = rmax16(s);
        float p = __expf(s - mx);
        prb[i] = p / rsum16(p);
    }
    // P -> A layout (K padded 16->32 with zeros)
#pragma unroll
    for (int i = 0; i < 8; ++i) {
        sb[(i + hf * 8) * 32 + m] = f2bf(prb[i]);
        sb[(i + hf * 8) * 32 + 16 + m] = 0;
    }
    __syncthreads();
    U16 pa;
#pragma unroll
    for (int v = 0; v < 8; ++v)
        pa.u[v] = *(const unsigned int*)(sb + m * 32 + KP(v, hf));
    __syncthreads();

    // pat_av = P @ pv  (4 tiles of 16 channels)
    v8f pav[4];
#pragma unroll
    for (int t4 = 0; t4 < 4; ++t4) {
        U16 bb;
#pragma unroll
        for (int v = 0; v < 8; ++v)
            bb.u[v] = (v < 4)
                ? *(const unsigned int*)(PVT + (size_t)(t4 * 16 + m) * 16 + hf * 8 + ((v & 3) << 1))
                : 0u;
        v8f c = {};
        pav[t4] = wmma_bf16(pa, bb, c);
    }

    // ---- main 24 output tiles (384 channels) ----
    for (int tile = 0; tile < 24; ++tile) {
        v8f c = {};
#pragma unroll
        for (int ks = 0; ks < 4; ++ks) {
            U16 bb; loadW(bb, Wall, tile * 16, ks, lane);
            c = wmma_bf16(ax[ks], bb, c);
        }
        if (tile >= 10 && tile < 14) {  // geo_k tiles: add pattern attention output
            v8f& add = pav[tile - 10];
#pragma unroll
            for (int i = 0; i < 8; ++i) c[i] += add[i];
        }
        // scatter to staged branch layouts
        const int e = tile * 16 + m;    // global output channel 0..383
#pragma unroll
        for (int i = 0; i < 8; ++i) {
            int rg = row0 + i + hf * 8;
            int n = rg & 511;
            int bt = rg >> 9;
            unsigned short val = f2bf(c[i]);
            if (e < 96) {               // temporal [b][n][h][t(pad16)][d]
                int sub = e / 32, cc = e % 32, hh2 = cc >> 4, d = cc & 15;
                unsigned short* dp = (sub == 0) ? TQ : (sub == 1) ? TK : TV;
                dp[(((size_t)(bt / 12) * 512 + n) * 2 + hh2) * 256 + (bt % 12) * 16 + d] = val;
            } else if (e < 288) {       // geo [b][t][h][n][d]
                int ee = e - 96; int sub = ee / 64, cc = ee % 64, hh2 = cc >> 4, d = cc & 15;
                unsigned short* dp = (sub == 0) ? GQ : (sub == 1) ? GK : GV;
                dp[(((size_t)bt * 4 + hh2) * 512 + n) * 16 + d] = val;
            } else {                    // sem [b][t][h][n][d]
                int ee = e - 288; int sub = ee / 32, cc = ee % 32, hh2 = cc >> 4, d = cc & 15;
                unsigned short* dp = (sub == 0) ? SQ : (sub == 1) ? SK : SV;
                dp[(((size_t)bt * 2 + hh2) * 512 + n) * 16 + d] = val;
            }
        }
    }
}

// ---------------------------------------------------------------------------
// Kernel 2: temporal attention, one wave per (b, n, h). T=12 padded to 16.
// ---------------------------------------------------------------------------
__global__ __launch_bounds__(32) void k2_temporal(
    const unsigned short* __restrict__ TQ, const unsigned short* __restrict__ TK,
    const unsigned short* __restrict__ TV, unsigned short* __restrict__ CAT) {
    __shared__ __align__(16) unsigned short sp[16 * 32];
    __shared__ __align__(16) unsigned short vt[16 * 16];

    const int lane = threadIdx.x;
    const int m = lane & 15, hf = lane >> 4;
    const int bid = blockIdx.x;
    const int h = bid & 1, n = (bid >> 1) & 511, b = bid >> 10;
    const size_t base = (((size_t)b * 512 + n) * 2 + h) * 256;  // 16 rows * 16

    U16 qa, ka;
    loadT16pad(qa, TQ + base, lane);
    loadT16pad(ka, TK + base, lane);
    v8f z = {};
    v8f s = wmma_bf16(qa, ka, z);

    v8f pr;
#pragma unroll
    for (int i = 0; i < 8; ++i) {
        float sv = s[i] * SCALE;
        if (m >= 12) sv = NEGINF;                  // mask padded key columns
        float mx = rmax16(sv);
        float p = (sv == NEGINF) ? 0.f : __expf(sv - mx);
        pr[i] = p / rsum16(p);
    }
#pragma unroll
    for (int i = 0; i < 8; ++i) {
        sp[(i + hf * 8) * 32 + m] = f2bf(pr[i]);
        sp[(i + hf * 8) * 32 + 16 + m] = 0;
    }
    {   // V^T into LDS: this lane covers key=m, channels hf*8..hf*8+7
        const unsigned short* vr = TV + base + (size_t)m * 16 + hf * 8;
#pragma unroll
        for (int j = 0; j < 4; ++j) {
            unsigned int vv = *(const unsigned int*)(vr + j * 2);
            vt[(hf * 8 + 2 * j) * 16 + m]     = (unsigned short)(vv & 0xffffu);
            vt[(hf * 8 + 2 * j + 1) * 16 + m] = (unsigned short)(vv >> 16);
        }
    }
    __syncthreads();
    U16 pa, vb;
#pragma unroll
    for (int v = 0; v < 8; ++v) {
        pa.u[v] = *(const unsigned int*)(sp + m * 32 + KP(v, hf));
        vb.u[v] = (v < 4) ? *(const unsigned int*)(vt + m * 16 + hf * 8 + ((v & 3) << 1)) : 0u;
    }
    v8f o = wmma_bf16(pa, vb, z);
#pragma unroll
    for (int i = 0; i < 8; ++i) {
        int t = i + hf * 8;
        if (t < 12)
            CAT[(((size_t)b * 12 + t) * 512 + n) * 128 + h * 16 + m] = f2bf(o[i]);
    }
}

// ---------------------------------------------------------------------------
// Kernel 3: spatial attention (geo: 4 heads w/ adj bias + mask, sem: 2 heads w/ mask).
// One wave per (b,t,head,qtile). Flash loop: 16 blocks of 32 keys.
// ---------------------------------------------------------------------------
__global__ __launch_bounds__(32) void k3_spatial(
    const unsigned short* __restrict__ GQ, const unsigned short* __restrict__ GK,
    const unsigned short* __restrict__ GV, const unsigned short* __restrict__ SQ,
    const unsigned short* __restrict__ SK, const unsigned short* __restrict__ SV,
    const float* __restrict__ adj, const unsigned char* __restrict__ gmask,
    const unsigned char* __restrict__ smask, unsigned short* __restrict__ CAT) {
    __shared__ __align__(16) unsigned short sp[16 * 32];
    __shared__ __align__(16) unsigned short vt[16 * 32];

    const int lane = threadIdx.x;
    const int m = lane & 15, hf = lane >> 4;
    const int bid = blockIdx.x;
    const int qi = bid & 31;
    const int r = bid >> 5;
    const int hh = r % 6;
    const int bt = r / 6;
    const int b = bt / 12;
    const bool geo = hh < 4;
    const int h = geo ? hh : hh - 4;

    const unsigned short* Qp = geo ? GQ + ((size_t)bt * 4 + h) * 512 * 16
                                   : SQ + ((size_t)bt * 2 + h) * 512 * 16;
    const unsigned short* Kp = geo ? GK + ((size_t)bt * 4 + h) * 512 * 16
                                   : SK + ((size_t)bt * 2 + h) * 512 * 16;
    const unsigned short* Vp = geo ? GV + ((size_t)bt * 4 + h) * 512 * 16
                                   : SV + ((size_t)bt * 2 + h) * 512 * 16;
    const unsigned char* msk = geo ? gmask : smask;

    U16 qa;
    loadT16pad(qa, Qp + (size_t)qi * 16 * 16, lane);

    float mi[8], li[8];
    v8f acc = {};
#pragma unroll
    for (int i = 0; i < 8; ++i) { mi[i] = NEGINF; li[i] = 0.f; }

    for (int kb = 0; kb < 16; ++kb) {
        const int kbase = kb * 32;
        if (kb < 15) {
            __builtin_prefetch(Kp + (size_t)(kbase + 32) * 16, 0, 0);
            __builtin_prefetch(Vp + (size_t)(kbase + 32) * 16, 0, 0);
        }
        U16 kb0, kb1;
        loadT16pad(kb0, Kp + (size_t)kbase * 16, lane);
        loadT16pad(kb1, Kp + (size_t)(kbase + 16) * 16, lane);
        v8f z = {};
        v8f s0 = wmma_bf16(qa, kb0, z);
        v8f s1 = wmma_bf16(qa, kb1, z);

        float p0[8], p1[8];
#pragma unroll
        for (int i = 0; i < 8; ++i) {
            const int q = qi * 16 + i + hf * 8;
            const int kc0 = kbase + m, kc1 = kbase + 16 + m;
            float v0 = s0[i] * SCALE, v1 = s1[i] * SCALE;
            if (geo) {
                const float* ar = adj + ((size_t)b * 512 + q) * 512;
                v0 += ar[kc0];
                v1 += ar[kc1];
            }
            if (msk[(size_t)q * 512 + kc0]) v0 = NEGINF;
            if (msk[(size_t)q * 512 + kc1]) v1 = NEGINF;
            float mloc = rmax16(fmaxf(v0, v1));
            float mnew = fmaxf(mi[i], mloc);
            float scl = (mi[i] == NEGINF) ? 0.f : __expf(mi[i] - mnew);
            float e0 = (v0 == NEGINF) ? 0.f : __expf(v0 - mnew);
            float e1 = (v1 == NEGINF) ? 0.f : __expf(v1 - mnew);
            li[i] = li[i] * scl + rsum16(e0 + e1);
            acc[i] *= scl;
            mi[i] = mnew;
            p0[i] = e0;
            p1[i] = e1;
        }

        __syncthreads();  // LDS buffers reusable (previous reads already in regs)
#pragma unroll
        for (int i = 0; i < 8; ++i) {
            sp[(i + hf * 8) * 32 + m] = f2bf(p0[i]);
            sp[(i + hf * 8) * 32 + 16 + m] = f2bf(p1[i]);
        }
        {   // V^T: this lane covers key kbase+lane (16 channels)
            const unsigned short* vr = Vp + (size_t)(kbase + lane) * 16;
#pragma unroll
            for (int j = 0; j < 8; ++j) {
                unsigned int vv = *(const unsigned int*)(vr + j * 2);
                vt[(2 * j) * 32 + lane]     = (unsigned short)(vv & 0xffffu);
                vt[(2 * j + 1) * 32 + lane] = (unsigned short)(vv >> 16);
            }
        }
        __syncthreads();
        U16 pa, vb;
#pragma unroll
        for (int v = 0; v < 8; ++v) {
            pa.u[v] = *(const unsigned int*)(sp + m * 32 + KP(v, hf));
            vb.u[v] = *(const unsigned int*)(vt + m * 32 + KP(v, hf));
        }
        acc = wmma_bf16(pa, vb, acc);
    }

    const int chb = (geo ? 32 + h * 16 : 96 + h * 16) + m;
#pragma unroll
    for (int i = 0; i < 8; ++i) {
        int q = qi * 16 + i + hf * 8;
        CAT[((size_t)bt * 512 + q) * 128 + chb] = f2bf(acc[i] / li[i]);
    }
}

// ---------------------------------------------------------------------------
// Kernel 4: output projection out = CAT @ proj_w^T + proj_b  (f32 out)
// ---------------------------------------------------------------------------
__global__ __launch_bounds__(32) void k4_out(
    const unsigned short* __restrict__ CAT, const unsigned short* __restrict__ WP,
    const float* __restrict__ proj_b, float* __restrict__ out) {
    const int lane = threadIdx.x;
    const int m = lane & 15, hf = lane >> 4;
    const int row0 = blockIdx.x * 16;

    U16 a[4];
#pragma unroll
    for (int ks = 0; ks < 4; ++ks)
#pragma unroll
        for (int v = 0; v < 8; ++v)
            a[ks].u[v] = *(const unsigned int*)(CAT + (size_t)(row0 + m) * 128 + ks * 32 + KP(v, hf));

    for (int nt = 0; nt < 8; ++nt) {
        v8f c = {};
#pragma unroll
        for (int ks = 0; ks < 4; ++ks) {
            U16 bb;
#pragma unroll
            for (int v = 0; v < 8; ++v)
                bb.u[v] = *(const unsigned int*)(WP + (size_t)(nt * 16 + m) * 128 + ks * 32 + KP(v, hf));
            c = wmma_bf16(a[ks], bb, c);
        }
        float bia = proj_b[nt * 16 + m];
#pragma unroll
        for (int i = 0; i < 8; ++i)
            out[(size_t)(row0 + i + hf * 8) * 128 + nt * 16 + m] = c[i] + bia;
    }
}

// ---------------------------------------------------------------------------
// Host launcher
// ---------------------------------------------------------------------------
extern "C" void kernel_launch(void* const* d_in, const int* in_sizes, int n_in,
                              void* d_out, int out_size, void* d_ws, size_t ws_size,
                              hipStream_t stream) {
    const float* x      = (const float*)d_in[0];
    const float* xp     = (const float*)d_in[1];
    const float* pkeys  = (const float*)d_in[2];
    const float* adj    = (const float*)d_in[3];
    const unsigned char* gmask = (const unsigned char*)d_in[4];
    const unsigned char* smask = (const unsigned char*)d_in[5];
    const float* t_qw   = (const float*)d_in[6];
    const float* t_kw   = (const float*)d_in[7];
    const float* t_vw   = (const float*)d_in[8];
    const float* geo_qw = (const float*)d_in[9];
    const float* geo_kw = (const float*)d_in[10];
    const float* geo_vw = (const float*)d_in[11];
    const float* sem_qw = (const float*)d_in[12];
    const float* sem_kw = (const float*)d_in[13];
    const float* sem_vw = (const float*)d_in[14];
    const float* pat_qw = (const float*)d_in[15];
    const float* pat_qb = (const float*)d_in[16];
    const float* pat_kw = (const float*)d_in[17];
    const float* pat_kb = (const float*)d_in[18];
    const float* pat_vw = (const float*)d_in[19];
    const float* pat_vb = (const float*)d_in[20];
    const float* proj_w = (const float*)d_in[21];
    const float* proj_b = (const float*)d_in[22];
    float* out = (float*)d_out;

    unsigned short* ws = (unsigned short*)d_ws;
    unsigned short* Wall = ws + OFF_W;
    unsigned short* PK   = ws + OFF_PK;
    unsigned short* PVT  = ws + OFF_PVT;
    unsigned short* TQ   = ws + OFF_TQ;
    unsigned short* TK   = ws + OFF_TK;
    unsigned short* TV   = ws + OFF_TV;
    unsigned short* GQ   = ws + OFF_GQ;
    unsigned short* GK   = ws + OFF_GK;
    unsigned short* GV   = ws + OFF_GV;
    unsigned short* SQ   = ws + OFF_SQ;
    unsigned short* SK   = ws + OFF_SK;
    unsigned short* SV   = ws + OFF_SV;
    unsigned short* CAT  = ws + OFF_CAT;

    // Stage 0: weights -> bf16, pattern pk/pv^T, clear padded temporal buffers
    k0_weights<<<(int)((NW + 255) / 256), 256, 0, stream>>>(
        Wall, t_qw, t_kw, t_vw, geo_qw, geo_kw, geo_vw,
        sem_qw, sem_kw, sem_vw, pat_qw, proj_w);
    k0_pkpv<<<1, 256, 0, stream>>>(PK, PVT, pkeys, pat_kw, pat_kb, pat_vw, pat_vb);
    {
        int n4 = (int)(3 * TSZ * sizeof(unsigned short) / sizeof(float4));
        k0_clear<<<(n4 + 255) / 256, 256, 0, stream>>>((float4*)TQ, n4);
    }

    // Stage 1: projections + pattern attention fused into geo_k
    k1_proj<<<BTN / 16, 32, 0, stream>>>(x, xp, Wall, PK, PVT, pat_qb,
                                         TQ, TK, TV, GQ, GK, GV, SQ, SK, SV);

    // Stage 2: temporal attention
    k2_temporal<<<Bc * Nc * 2, 32, 0, stream>>>(TQ, TK, TV, CAT);

    // Stage 3: spatial (geo + sem) flash attention
    k3_spatial<<<Bc * Tc * 6 * (Nc / 16), 32, 0, stream>>>(
        GQ, GK, GV, SQ, SK, SV, adj, gmask, smask, CAT);

    // Stage 4: output projection
    k4_out<<<BTN / 16, 32, 0, stream>>>(CAT, Wall + (size_t)448 * 128, proj_b, out);
}
